// TypedBinaryTreeLSTM_41497974014351
// MI455X (gfx1250) — matrix operand order repaired and checked by reading, not compiled, not run on hardware
//
#include <hip/hip_runtime.h>
#include <hip/hip_bf16.h>

typedef __attribute__((ext_vector_type(16))) _Float16 v16h;
typedef __attribute__((ext_vector_type(8)))  float    v8f;

#define Bn   256
#define Nn   16
#define Mn   128
#define Vv   64
#define Hn   128
#define Kn   8
#define NP1  17
#define KT   136   // K*(N+1)
#define NTYPE 15

// ---------------------------------------------------------------------------
// Kernel A: per-type logit table via WMMA.
//   table[t][k] = sum_h w_sem[t,k,h] * type_emb[t+9,h] + b_sem[t,k]
// One wave (32 threads) per (type, 16-row k-block). A = w rows (f16), B = emb
// chunk broadcast to all 16 columns, so every column of C is the GEMV result.
// ---------------------------------------------------------------------------
__global__ __launch_bounds__(32)
void build_table_wmma(const float* __restrict__ type_emb,
                      const float* __restrict__ w_sem,
                      const float* __restrict__ b_sem,
                      float* __restrict__ table) {
  const int t    = blockIdx.x / 9;
  const int kb   = blockIdx.x % 9;
  const int k0   = kb * 16;
  const int lane = threadIdx.x;      // 0..31
  const int mrow  = lane & 15;       // tile row M
  const int khalf = lane >> 4;       // 0: K 0..7 & 16..23 ; 1: K 8..15 & 24..31

  const float* emb = type_emb + (size_t)(t + 9) * Hn;
  const float* w   = w_sem + (size_t)t * KT * Hn;

  v8f c = {};
  for (int h0 = 0; h0 < Hn; h0 += 32) {
    v16h a, bf;
    const int krow = k0 + mrow;
    if (krow < KT) {
      const float* wrow = w + (size_t)krow * Hn + h0;
      #pragma unroll
      for (int i = 0; i < 8; ++i) {
        a[i]     = (_Float16)wrow[khalf * 8 + i];        // K = khalf*8 + i
        a[8 + i] = (_Float16)wrow[16 + khalf * 8 + i];   // K = 16 + khalf*8 + i
      }
    } else {
      #pragma unroll
      for (int i = 0; i < 16; ++i) a[i] = (_Float16)0.0f;
    }
    // B[32x16]: lane holds K = khalf*8+i (i<8), 16+khalf*8+(i-8) (i>=8) for column N=lane%16; all
    // columns identical = emb chunk -> every C column equals the GEMV.
    #pragma unroll
    for (int i = 0; i < 16; ++i)
      bf[i] = (_Float16)emb[h0 + (i < 8 ? khalf * 8 + i : 16 + khalf * 8 + (i - 8))];

    c = __builtin_amdgcn_wmma_f32_16x16x32_f16(false, a, false, bf,
                                               (short)0, c, false, false);
  }

  // C layout: VGPR r, lane L -> row r + 8*(L/16), col L%16. Column-0 holders
  // (lanes 0 and 16) write the result.
  if ((lane & 15) == 0) {
    const int rbase = khalf * 8;
    #pragma unroll
    for (int r = 0; r < 8; ++r) {
      const int k = k0 + rbase + r;
      if (k < KT) table[t * KT + k] = c[r] + b_sem[t * KT + k];
    }
  }
}

// ---------------------------------------------------------------------------
// Kernel B: per-example gumbel argmax selection, length scan, packing.
// 256 threads = 8 waves per example; wave k owns template k.
// ---------------------------------------------------------------------------
__global__ __launch_bounds__(256)
void pack_templates(const float* __restrict__ decodings,
                    const float* __restrict__ gumbel,
                    const int*   __restrict__ target_types,
                    const int*   __restrict__ spans,
                    const float* __restrict__ table,
                    float* __restrict__ out) {
  __shared__ int s_sel[Kn];
  __shared__ int s_len[Kn];
  __shared__ int s_off[Kn];
  __shared__ int s_eff[Kn];

  const int b   = blockIdx.x;
  const int tid = threadIdx.x;
  const int t   = target_types[b];

  // Phase 1: sel[k] = argmax_n (table[t-9,k,n] + gumbel[b,k,n]) over n<=span.
  // type==20 uses the fixed start template: row0 -> child 1, rest -> pad(0).
  if (tid < Kn) {
    const int k = tid;
    int s;
    if (t == 20) {
      s = (k == 0) ? 1 : 0;
    } else {
      const float* tab = table + (size_t)(t - 9) * KT + k * NP1;
      const float* g   = gumbel + ((size_t)b * Kn + k) * NP1;
      const int span   = spans[b];
      float best = -3.4e38f; int bi = 0;
      for (int n = 0; n < NP1; ++n) {
        if (n <= span) {
          const float v = tab[n] + g[n];
          if (v > best) { best = v; bi = n; }   // first-max tie-break
        }
      }
      s = bi;
    }
    s_sel[k] = s;
  }
  __syncthreads();

  // Phase 2: lens[k] = last m+1 where argmax_v(decodings[b,sel-1,m,:]) != 0.
  // argmax != 0  <=>  max over v>0 strictly exceeds value[0] (first-tie).
  const int k    = tid >> 5;
  const int lane = tid & 31;
  const int s    = s_sel[k];
  int mylen = 0;
  if (s > 0) {
    const float* src = decodings + (((size_t)b * Nn) + (s - 1)) * (Mn * Vv);
    for (int m = lane; m < Mn; m += 32) {
      const float4* row = (const float4*)(src + m * Vv);
      const float4 r0 = row[0];
      const float first = r0.x;
      float mx = fmaxf(r0.y, fmaxf(r0.z, r0.w));
      #pragma unroll
      for (int i = 1; i < 16; ++i) {
        const float4 rr = row[i];
        mx = fmaxf(mx, fmaxf(fmaxf(rr.x, rr.y), fmaxf(rr.z, rr.w)));
      }
      if (mx > first) mylen = m + 1;   // m increases -> last hit is the max
    }
  }
  #pragma unroll
  for (int off = 16; off > 0; off >>= 1)
    mylen = max(mylen, __shfl_down(mylen, off, 32));
  if (lane == 0) s_len[k] = mylen;
  __syncthreads();

  // Phase 3: sequential packing scan (K=8, trivial).
  if (tid == 0) {
    int idx = 0;
    for (int kk = 0; kk < Kn; ++kk) {
      const int eff = min(s_len[kk], Mn - idx);
      s_off[kk] = idx; s_eff[kk] = eff; idx += eff;
    }
  }
  __syncthreads();

  // Phase 4: each output row p has exactly one source (or zero). 2 threads
  // per row, each moves 8 float4 (32 floats of V=64).
  const int p    = tid >> 1;
  const int half = tid & 1;
  int srck = -1;
  #pragma unroll
  for (int kk = 0; kk < Kn; ++kk)
    if (p >= s_off[kk] && p < s_off[kk] + s_eff[kk]) srck = kk;

  float4* dst = (float4*)(out + (((size_t)b * Mn) + p) * Vv) + half * 8;
  if (srck >= 0) {
    const int n = s_sel[srck] - 1;
    const int j = p - s_off[srck];
    const float4* src =
        (const float4*)(decodings + ((((size_t)b * Nn) + n) * Mn + j) * Vv) + half * 8;
    #pragma unroll
    for (int i = 0; i < 8; ++i) dst[i] = src[i];
  } else {
    const float4 z = {0.0f, 0.0f, 0.0f, 0.0f};
    #pragma unroll
    for (int i = 0; i < 8; ++i) dst[i] = z;
  }
}

extern "C" void kernel_launch(void* const* d_in, const int* in_sizes, int n_in,
                              void* d_out, int out_size, void* d_ws, size_t ws_size,
                              hipStream_t stream) {
  (void)in_sizes; (void)n_in; (void)out_size; (void)ws_size;
  const float* decodings    = (const float*)d_in[0]; // [256,16,128,64]
  const float* type_emb     = (const float*)d_in[1]; // [24,128]
  const float* w_sem        = (const float*)d_in[2]; // [15,136,128]
  const float* b_sem        = (const float*)d_in[3]; // [15,136]
  const float* gumbel       = (const float*)d_in[4]; // [256,8,17]
  const int*   target_types = (const int*)d_in[5];   // [256]
  const int*   spans        = (const int*)d_in[6];   // [256]
  float* out   = (float*)d_out;                      // [256,128,64]
  float* table = (float*)d_ws;                       // [15,136] scratch

  build_table_wmma<<<NTYPE * 9, 32, 0, stream>>>(type_emb, w_sem, b_sem, table);
  pack_templates<<<Bn, 256, 0, stream>>>(decodings, gumbel, target_types, spans,
                                         table, out);
}
